// DotPructAttention_7438883357023
// MI455X (gfx1250) — matrix operand order repaired and checked by reading, not compiled, not run on hardware
//
#include <hip/hip_runtime.h>
#include <hip/hip_bf16.h>

// DotProductAttention for MI455X (gfx1250), wave32 flash-attention with
// v_wmma_f32_16x16x32_f16. B*H=128, S=1024, D=64, fp32 in/out.
// Round 4: softmax denominator computed on the XDL pipe via a ones-column
// P*V WMMA (no DPP sum reduction), scalar-threshold masking, 64-key tiles,
// mod-4 key interleave, DPP max reduction, log2-domain online softmax.

typedef __attribute__((ext_vector_type(16))) _Float16 v16h;
typedef __attribute__((ext_vector_type(8)))  _Float16 v8h;
typedef __attribute__((ext_vector_type(4)))  _Float16 v4h;
typedef __attribute__((ext_vector_type(8)))  float    v8f;

#define WMMA_F16(A, B, C) \
    __builtin_amdgcn_wmma_f32_16x16x32_f16(false, (A), false, (B), (short)0, (C), false, false)

// DPP butterfly within each 16-lane half:
// 0xB1 quad_perm(1,0,3,2)=xor1 ; 0x4E quad_perm(2,3,0,1)=xor2
// 0x141 row_half_mirror (4-group swap) ; 0x140 row_mirror (8-group swap)
template <int CTRL>
__device__ __forceinline__ float dpp_swap(float v) {
    int i = __builtin_bit_cast(int, v);
    int r = __builtin_amdgcn_update_dpp(i, i, CTRL, 0xF, 0xF, true);
    return __builtin_bit_cast(float, r);
}
__device__ __forceinline__ float redmax16(float v) {
    v = fmaxf(v, dpp_swap<0xB1>(v));
    v = fmaxf(v, dpp_swap<0x4E>(v));
    v = fmaxf(v, dpp_swap<0x141>(v));
    v = fmaxf(v, dpp_swap<0x140>(v));
    return v;
}
__device__ __forceinline__ v16h cat16(v8h a, v8h b) {
    return __builtin_shufflevector(a, b, 0,1,2,3,4,5,6,7,8,9,10,11,12,13,14,15);
}

__global__ __launch_bounds__(256)
void attn_fa_wmma_kernel(const float* __restrict__ Q,
                         const float* __restrict__ K,
                         const float* __restrict__ V,
                         const int*   __restrict__ valid_lens,
                         float*       __restrict__ O) {
    constexpr int S = 1024, D = 64, H = 16, KT = 64;
    // scale (1/sqrt(64)) * log2(e): softmax runs in base-2 domain.
    constexpr float QS      = 0.125f * 1.44269504088896340736f;
    constexpr float MASKVAL = 1.4426950408889634e-6f;   // 1e-6 * log2(e)

    // K tile: row = permuted key slot (key%4)*16 + key/4, 64 d, stride 72
    __shared__ _Float16 sK[64][72];
    // V^T tile: [64 d][64 keys natural order], stride 72
    __shared__ _Float16 sVt[64][72];
    // per-wave P staging: [wave][16 rows][64 keys natural order]
    __shared__ _Float16 sP[8][16][64];

    const int tid  = threadIdx.x;
    const int lane = tid & 31;
    const int w    = tid >> 5;
    const int h    = lane >> 4;     // lane half (0/1)
    const int m16  = lane & 15;     // lane-within-half
    const int c4m  = 4 * m16;       // loop-invariant mask operand

    const int head  = blockIdx.y;   // 0..127
    const int q0    = blockIdx.x * 128 + w * 16;
    const int valid = valid_lens[head / H];

    const size_t hbase = (size_t)head * S * D;

    // ---- Load Q tile (16x64) into A-layout f16 registers, pre-scaled ----
    const float* gQ = Q + hbase + (size_t)(q0 + m16) * D;
    v16h qa0, qa1;
    {
        float4 f0 = *(const float4*)(gQ + 8*h);
        float4 f1 = *(const float4*)(gQ + 8*h + 4);
        float4 f2 = *(const float4*)(gQ + 8*h + 16);
        float4 f3 = *(const float4*)(gQ + 8*h + 20);
        qa0 = (v16h){(_Float16)(f0.x*QS),(_Float16)(f0.y*QS),(_Float16)(f0.z*QS),(_Float16)(f0.w*QS),
                     (_Float16)(f1.x*QS),(_Float16)(f1.y*QS),(_Float16)(f1.z*QS),(_Float16)(f1.w*QS),
                     (_Float16)(f2.x*QS),(_Float16)(f2.y*QS),(_Float16)(f2.z*QS),(_Float16)(f2.w*QS),
                     (_Float16)(f3.x*QS),(_Float16)(f3.y*QS),(_Float16)(f3.z*QS),(_Float16)(f3.w*QS)};
        f0 = *(const float4*)(gQ + 32 + 8*h);
        f1 = *(const float4*)(gQ + 32 + 8*h + 4);
        f2 = *(const float4*)(gQ + 32 + 8*h + 16);
        f3 = *(const float4*)(gQ + 32 + 8*h + 20);
        qa1 = (v16h){(_Float16)(f0.x*QS),(_Float16)(f0.y*QS),(_Float16)(f0.z*QS),(_Float16)(f0.w*QS),
                     (_Float16)(f1.x*QS),(_Float16)(f1.y*QS),(_Float16)(f1.z*QS),(_Float16)(f1.w*QS),
                     (_Float16)(f2.x*QS),(_Float16)(f2.y*QS),(_Float16)(f2.z*QS),(_Float16)(f2.w*QS),
                     (_Float16)(f3.x*QS),(_Float16)(f3.y*QS),(_Float16)(f3.z*QS),(_Float16)(f3.w*QS)};
    }

    // all-ones B operand for the denominator WMMA
    const v16h ones = {(_Float16)1.f,(_Float16)1.f,(_Float16)1.f,(_Float16)1.f,
                       (_Float16)1.f,(_Float16)1.f,(_Float16)1.f,(_Float16)1.f,
                       (_Float16)1.f,(_Float16)1.f,(_Float16)1.f,(_Float16)1.f,
                       (_Float16)1.f,(_Float16)1.f,(_Float16)1.f,(_Float16)1.f};

    v8f acc[4] = {};
    v8f accl   = {};                 // row-sum accumulator (softmax denominator)
    float mrow[8];
    #pragma unroll
    for (int r = 0; r < 8; ++r) mrow[r] = -3.0e38f;
    const v8f zf = {};

    for (int kt = 0; kt < S; kt += KT) {
        __syncthreads();  // protect sK/sVt from previous-iteration readers

        // ---- Cooperative stage: K (mod-4 row permute) and V (4x4 transpose) ----
        const float4* gK4 = (const float4*)(K + hbase + (size_t)kt * D);
        const float4* gV4 = (const float4*)(V + hbase + (size_t)kt * D);
        #pragma unroll
        for (int it = 0; it < 4; ++it) {
            int i   = tid + 256 * it;      // 0..1023 float4s = 64x64 floats
            int row = i >> 4;              // key 0..63
            int c4  = (i & 15) << 2;       // d
            float4 kx = gK4[i];
            v4h kh = {(_Float16)kx.x, (_Float16)kx.y, (_Float16)kx.z, (_Float16)kx.w};
            int prow = ((row & 3) << 4) | (row >> 2);   // subtile interleave
            *(v4h*)(&sK[prow][c4]) = kh;
        }
        {   // V: each thread transposes a 4x4 block -> 4 aligned b64 stores
            int kb4 = (tid & 15) << 2;     // key base
            int db4 = (tid >> 4) << 2;     // d base
            float4 r0 = gV4[(kb4 + 0) * 16 + (db4 >> 2)];
            float4 r1 = gV4[(kb4 + 1) * 16 + (db4 >> 2)];
            float4 r2 = gV4[(kb4 + 2) * 16 + (db4 >> 2)];
            float4 r3 = gV4[(kb4 + 3) * 16 + (db4 >> 2)];
            v4h c0 = {(_Float16)r0.x, (_Float16)r1.x, (_Float16)r2.x, (_Float16)r3.x};
            v4h c1 = {(_Float16)r0.y, (_Float16)r1.y, (_Float16)r2.y, (_Float16)r3.y};
            v4h c2 = {(_Float16)r0.z, (_Float16)r1.z, (_Float16)r2.z, (_Float16)r3.z};
            v4h c3 = {(_Float16)r0.w, (_Float16)r1.w, (_Float16)r2.w, (_Float16)r3.w};
            *(v4h*)(&sVt[db4 + 0][kb4]) = c0;
            *(v4h*)(&sVt[db4 + 1][kb4]) = c1;
            *(v4h*)(&sVt[db4 + 2][kb4]) = c2;
            *(v4h*)(&sVt[db4 + 3][kb4]) = c3;
        }
        if (kt + KT < S) {   // CDNA5 global_prefetch_b8 for next tile
            __builtin_prefetch((const void*)(gK4 + 1024 + tid * 4), 0, 0);
            __builtin_prefetch((const void*)(gV4 + 1024 + tid * 4), 0, 0);
        }
        __syncthreads();

        // ---- QK^T: 4 score subtiles (subtile ks = keys kt + 4*j + ks) ----
        v8f s[4];
        #pragma unroll
        for (int ks = 0; ks < 4; ++ks) {
            const v8h* pb0 = (const v8h*)&sK[ks * 16 + m16][8 * h];        // d 0..31
            const v8h* pb1 = (const v8h*)&sK[ks * 16 + m16][32 + 8 * h];   // d 32..63
            v16h b0 = cat16(pb0[0], pb0[2]);
            v16h b1 = cat16(pb1[0], pb1[2]);
            v8f t = WMMA_F16(qa0, b0, zf);
            s[ks]  = WMMA_F16(qa1, b1, t);
        }

        // ---- masked_fill + online softmax (log2 domain) ----
        // mask: (kt + 4*m16 + ks) >= valid  <=>  4*m16 >= (valid - kt) - ks
        const int thr = valid - kt;
        const bool mk0 = c4m >= thr;
        const bool mk1 = c4m >= thr - 1;
        const bool mk2 = c4m >= thr - 2;
        const bool mk3 = c4m >= thr - 3;
        float alpha[8];
        #pragma unroll
        for (int r = 0; r < 8; ++r) {
            float x0 = mk0 ? MASKVAL : s[0][r];
            float x1 = mk1 ? MASKVAL : s[1][r];
            float x2 = mk2 ? MASKVAL : s[2][r];
            float x3 = mk3 ? MASKVAL : s[3][r];
            float mx   = redmax16(fmaxf(fmaxf(x0, x1), fmaxf(x2, x3)));
            float mnew = fmaxf(mrow[r], mx);
            alpha[r] = __builtin_amdgcn_exp2f(mrow[r] - mnew);
            mrow[r]  = mnew;
            float e0 = __builtin_amdgcn_exp2f(x0 - mnew);
            float e1 = __builtin_amdgcn_exp2f(x1 - mnew);
            float e2 = __builtin_amdgcn_exp2f(x2 - mnew);
            float e3 = __builtin_amdgcn_exp2f(x3 - mnew);
            // packed P store: keys colb..colb+3 are adjacent in natural order
            v4h ph = {(_Float16)e0, (_Float16)e1, (_Float16)e2, (_Float16)e3};
            *(v4h*)(&sP[w][r + 8 * h][4 * m16]) = ph;
        }
        #pragma unroll
        for (int n = 0; n < 4; ++n)
            #pragma unroll
            for (int r = 0; r < 8; ++r) acc[n][r] *= alpha[r];
        #pragma unroll
        for (int r = 0; r < 8; ++r) accl[r] *= alpha[r];

        __builtin_amdgcn_wave_barrier();   // DS pipe is in-order within a wave

        // ---- P·V (+ ones column for denominator): 10 WMMAs ----
        v16h pa[2];
        #pragma unroll
        for (int c = 0; c < 2; ++c) {
            const v8h* pp0 = (const v8h*)&sP[w][m16][c * 32 + 8 * h];
            const v8h* pp1 = (const v8h*)&sP[w][m16][c * 32 + 16 + 8 * h];
            pa[c] = cat16(pp0[0], pp1[0]);
        }
        #pragma unroll
        for (int n = 0; n < 4; ++n) {
            #pragma unroll
            for (int c = 0; c < 2; ++c) {
                const v8h* pv = (const v8h*)&sVt[n * 16 + m16][c * 32 + 8 * h];
                v16h vb = cat16(pv[0], pv[2]);
                acc[n] = WMMA_F16(pa[c], vb, acc[n]);
            }
        }
        accl = WMMA_F16(pa[0], ones, accl);     // denominator on the XDL pipe
        accl = WMMA_F16(pa[1], ones, accl);
    }

    // ---- normalize and write output (fp32) ----
    float invl[8];
    #pragma unroll
    for (int r = 0; r < 8; ++r) invl[r] = __builtin_amdgcn_rcpf(accl[r]);
    #pragma unroll
    for (int n = 0; n < 4; ++n) {
        #pragma unroll
        for (int r = 0; r < 8; ++r) {
            int row = q0 + r + 8 * h;
            O[hbase + (size_t)row * D + n * 16 + m16] = acc[n][r] * invl[r];
        }
    }
}

extern "C" void kernel_launch(void* const* d_in, const int* in_sizes, int n_in,
                              void* d_out, int out_size, void* d_ws, size_t ws_size,
                              hipStream_t stream) {
    (void)in_sizes; (void)n_in; (void)out_size; (void)d_ws; (void)ws_size;
    const float* Q = (const float*)d_in[0];
    const float* K = (const float*)d_in[1];
    const float* V = (const float*)d_in[2];
    const int*   L = (const int*)d_in[3];
    float*       O = (float*)d_out;
    // grid: 8 query blocks (128 rows each) x 128 heads; 256 threads = 8 waves
    dim3 grid(1024 / 128, 8 * 16);
    attn_fa_wmma_kernel<<<grid, 256, 0, stream>>>(Q, K, V, L, O);
}
